// Attention_54391465837214
// MI455X (gfx1250) — matrix operand order repaired and checked by reading, not compiled, not run on hardware
//
#include <hip/hip_runtime.h>
#include <math.h>

// ---------------------------------------------------------------------------
// Problem constants: B=512, T=128, D=512, H=512, C=96, S=26, XK = D+C+H = 1120
// ---------------------------------------------------------------------------
#define BB 512
#define TT 128
#define DD 512
#define HH 512
#define CC 96
#define SS 26
#define XK 1120

// ---------------------------------------------------------------------------
// bf16 helpers (storage = unsigned short, RNE) + fast transcendentals
// ---------------------------------------------------------------------------
__device__ __forceinline__ unsigned short f32_to_bf16(float f) {
    unsigned u = __float_as_uint(f);
    u += 0x7FFFu + ((u >> 16) & 1u);
    return (unsigned short)(u >> 16);
}
__device__ __forceinline__ float bf16_to_f32(unsigned h) {
    return __uint_as_float(h << 16);
}
__device__ __forceinline__ unsigned pack_bf16x2(float lo, float hi) {
    return (unsigned)f32_to_bf16(lo) | ((unsigned)f32_to_bf16(hi) << 16);
}
// tanh(x) = (e^{2x}-1)/(e^{2x}+1), v_exp_f32 + v_rcp_f32 (4 VALU ops)
__device__ __forceinline__ float fast_tanh(float x) {
    x = fminf(fmaxf(x, -15.f), 15.f);
    const float e = __expf(2.f * x);
    return (e - 1.f) * __builtin_amdgcn_rcpf(e + 1.f);
}
__device__ __forceinline__ float fast_sigmoid(float x) {
    return __builtin_amdgcn_rcpf(1.f + __expf(-x));
}

// ---------------------------------------------------------------------------
// WMMA fragment plumbing (gfx1250 wave32, v_wmma_f32_16x16x32_bf16)
//   A 16x32 bf16 : lane {half,mr}; VGPR p holds K=(p<4?0:16)+8*half+2*(p&3),+1
//   B 32x16 bf16 : lane {half,nr}; VGPR p holds K=(p<4?0:16)+8*half+2*(p&3),+1  (B stored [N,K])
//   C/D 16x16 f32: VGPR i -> M=i+8*half, N=lane&15
// Row-major [M,K] / [N,K] makes each K-pair one aligned u32; the 8 u32 of a
// fragment coalesce into two global_load_b128.
// ---------------------------------------------------------------------------
typedef __attribute__((ext_vector_type(16))) __bf16 v16bf;
typedef __attribute__((ext_vector_type(8)))  float  v8f;

union FragBF { unsigned int u[8]; v16bf v; };

__device__ __forceinline__ FragBF load_a_frag(const unsigned short* __restrict__ A,
                                              int lda, int m0, int k0,
                                              unsigned half, unsigned lr) {
    FragBF f;
    const unsigned int* row =
        (const unsigned int*)(A + (size_t)(m0 + (int)lr) * lda + k0);
    const unsigned kb = half * 8u;
#pragma unroll
    for (int p = 0; p < 8; ++p) {
        unsigned k = ((p & 4) ? 16u : 0u) + kb + 2u * (unsigned)(p & 3);
        f.u[p] = row[k >> 1];
    }
    return f;
}

// Branchless: caller pre-clamps n0 so the load is always in-bounds.
// Out-of-tile fragments feed accumulators that are never stored.
__device__ __forceinline__ FragBF load_b_frag(const unsigned short* __restrict__ B,
                                              int ldb, int n0c, int k0,
                                              unsigned half, unsigned lr) {
    FragBF f;
    const unsigned int* row =
        (const unsigned int*)(B + (size_t)(n0c + (int)lr) * ldb + k0);
    const unsigned base = half * 4u;
#pragma unroll
    for (int p = 0; p < 8; ++p) f.u[p] = row[((p & 4) ? 8u : 0u) + base + (unsigned)(p & 3)];
    return f;
}

struct Frags {
    FragBF a0, a1, b0, b1, b2, b3;
};

__device__ __forceinline__ void store_tile(v8f acc, int m0, int n0,
                                           unsigned half, unsigned lr,
                                           float* C, int ldc,
                                           unsigned short* D16, int ldd,
                                           const float* bias, bool valid) {
    if (!valid) return;
    const int col = n0 + (int)lr;
    const float bv = bias ? bias[col] : 0.f;
#pragma unroll
    for (int i = 0; i < 8; ++i) {
        const int row = m0 + i + (int)half * 8;
        const float v = acc[i] + bv;
        if (C)   C[(size_t)row * ldc + col]   = v;
        if (D16) D16[(size_t)row * ldd + col] = f32_to_bf16(v);
    }
}

// ---------------------------------------------------------------------------
// GEMM: out[M,N] = A[M,K] * B[N,K]^T + bias[N];  A,B bf16 row-major.
// Block tile 128(M) x 128(N); 8 waves as 4(M) x 2(N); 32x64 per wave
// (2 A frags, 4 B frags, 8 f32 accumulators, 8 wmma per K-step).
// Register double-buffered K pipeline; fully branchless inner loop.
// Requires M%128==0, N%16==0, K%32==0.
// ---------------------------------------------------------------------------
__global__ __launch_bounds__(256)
void gemm_bf16_wmma_kernel(const unsigned short* __restrict__ A, int lda,
                           const unsigned short* __restrict__ Bm, int ldb,
                           float* __restrict__ C, int ldc,
                           unsigned short* __restrict__ D16, int ldd,
                           const float* __restrict__ bias,
                           int M, int N, int K) {
    const unsigned tid  = threadIdx.x;
    const unsigned wid  = tid >> 5;
    const unsigned lane = tid & 31u;
    const unsigned half = lane >> 4;
    const unsigned lr   = lane & 15u;

    const int m0 = (int)blockIdx.y * 128 + (int)(wid & 3u) * 32;
    const int n0 = (int)blockIdx.x * 128 + (int)(wid >> 2)  * 64;

    // store-side validity (wave-uniform); loads are clamped instead of guarded
    const bool ok0 = (n0 + 16) <= N;
    const bool ok1 = (n0 + 32) <= N;
    const bool ok2 = (n0 + 48) <= N;
    const bool ok3 = (n0 + 64) <= N;
    const int  nc0 = min(n0,      N - 16);
    const int  nc1 = min(n0 + 16, N - 16);
    const int  nc2 = min(n0 + 32, N - 16);
    const int  nc3 = min(n0 + 48, N - 16);

    v8f c00 = {}, c01 = {}, c02 = {}, c03 = {};
    v8f c10 = {}, c11 = {}, c12 = {}, c13 = {};

    auto load_all = [&](Frags& f, int k0) {
        f.a0 = load_a_frag(A, lda, m0,      k0, half, lr);
        f.a1 = load_a_frag(A, lda, m0 + 16, k0, half, lr);
        f.b0 = load_b_frag(Bm, ldb, nc0, k0, half, lr);
        f.b1 = load_b_frag(Bm, ldb, nc1, k0, half, lr);
        f.b2 = load_b_frag(Bm, ldb, nc2, k0, half, lr);
        f.b3 = load_b_frag(Bm, ldb, nc3, k0, half, lr);
    };
    auto mm = [&](const Frags& f) {
        c00 = __builtin_amdgcn_wmma_f32_16x16x32_bf16(false, f.a0.v, false, f.b0.v,
                                                      (short)0, c00, false, false);
        c01 = __builtin_amdgcn_wmma_f32_16x16x32_bf16(false, f.a0.v, false, f.b1.v,
                                                      (short)0, c01, false, false);
        c02 = __builtin_amdgcn_wmma_f32_16x16x32_bf16(false, f.a0.v, false, f.b2.v,
                                                      (short)0, c02, false, false);
        c03 = __builtin_amdgcn_wmma_f32_16x16x32_bf16(false, f.a0.v, false, f.b3.v,
                                                      (short)0, c03, false, false);
        c10 = __builtin_amdgcn_wmma_f32_16x16x32_bf16(false, f.a1.v, false, f.b0.v,
                                                      (short)0, c10, false, false);
        c11 = __builtin_amdgcn_wmma_f32_16x16x32_bf16(false, f.a1.v, false, f.b1.v,
                                                      (short)0, c11, false, false);
        c12 = __builtin_amdgcn_wmma_f32_16x16x32_bf16(false, f.a1.v, false, f.b2.v,
                                                      (short)0, c12, false, false);
        c13 = __builtin_amdgcn_wmma_f32_16x16x32_bf16(false, f.a1.v, false, f.b3.v,
                                                      (short)0, c13, false, false);
    };

    // software pipeline: loads for step k run ahead of wmma for step k-32
    Frags cur, nxt;
    load_all(cur, 0);
    for (int k0 = 32; k0 < K; k0 += 32) {
        load_all(nxt, k0);
        mm(cur);
        cur = nxt;
    }
    mm(cur);

    store_tile(c00, m0,      n0,      half, lr, C, ldc, D16, ldd, bias, ok0);
    store_tile(c01, m0,      n0 + 16, half, lr, C, ldc, D16, ldd, bias, ok1);
    store_tile(c02, m0,      n0 + 32, half, lr, C, ldc, D16, ldd, bias, ok2);
    store_tile(c03, m0,      n0 + 48, half, lr, C, ldc, D16, ldd, bias, ok3);
    store_tile(c10, m0 + 16, n0,      half, lr, C, ldc, D16, ldd, bias, ok0);
    store_tile(c11, m0 + 16, n0 + 16, half, lr, C, ldc, D16, ldd, bias, ok1);
    store_tile(c12, m0 + 16, n0 + 32, half, lr, C, ldc, D16, ldd, bias, ok2);
    store_tile(c13, m0 + 16, n0 + 48, half, lr, C, ldc, D16, ldd, bias, ok3);
}

// ---------------------------------------------------------------------------
// Fused attention: tanh-score -> softmax(T) -> context; one block per b.
// Bandwidth-critical: b128 loads on the score scan, u32 loads on the context
// scan, global_prefetch for the next score row.
// ---------------------------------------------------------------------------
__global__ __launch_bounds__(256)
void attn_ctx_kernel(const unsigned short* __restrict__ Hproj16, // [B*T, H]
                     const unsigned short* __restrict__ bH16,    // [B*T, D]
                     const float* __restrict__ hp,               // [B, H] (+b_h2h)
                     const float* __restrict__ w_score,          // [H]
                     const int* __restrict__ text,               // [B, S]
                     unsigned short* __restrict__ xcat,          // [B, XK]
                     int s) {
    __shared__ float hp_s[HH];
    __shared__ float ws_s[HH];
    __shared__ float e_s[TT];
    __shared__ float red_s[TT];

    const int b   = blockIdx.x;
    const int tid = threadIdx.x;

    for (int h = tid; h < HH; h += 256) {
        hp_s[h] = hp[(size_t)b * HH + h];
        ws_s[h] = w_score[h];
    }
    __syncthreads();

    // --- scores: e[t] = sum_h tanh(Hproj[b,t,h] + hp[h]) * w_score[h] ---
    const int wid  = tid >> 5;
    const int lane = tid & 31;
    for (int t = wid; t < TT; t += 8) {
        const uint4* row4 = (const uint4*)(Hproj16 + ((size_t)b * TT + t) * HH);
        if (t + 8 < TT) {
            const char* nrow = (const char*)(Hproj16 + ((size_t)b * TT + t + 8) * HH);
            __builtin_prefetch(nrow + lane * 32, 0, 0);   // cover the 1KB row
        }
        float sum = 0.f;
#pragma unroll
        for (int part = 0; part < 2; ++part) {
            const int idx = lane + 32 * part;           // uint4 index, 64 per row
            const uint4 q = row4[idx];
            const int hb = idx * 8;
            const unsigned w[4] = {q.x, q.y, q.z, q.w};
#pragma unroll
            for (int j = 0; j < 4; ++j) {
                const int h0 = hb + 2 * j;
                const float v0 = bf16_to_f32(w[j] & 0xFFFFu) + hp_s[h0];
                const float v1 = bf16_to_f32(w[j] >> 16)     + hp_s[h0 + 1];
                sum += fast_tanh(v0) * ws_s[h0];
                sum += fast_tanh(v1) * ws_s[h0 + 1];
            }
        }
        for (int off = 16; off > 0; off >>= 1) sum += __shfl_down(sum, off, 32);
        if (lane == 0) e_s[t] = sum;
    }
    __syncthreads();

    // --- softmax over T=128 ---
    if (tid < TT) red_s[tid] = e_s[tid];
    __syncthreads();
    for (int off = 64; off > 0; off >>= 1) {
        if (tid < off) red_s[tid] = fmaxf(red_s[tid], red_s[tid + off]);
        __syncthreads();
    }
    const float mx = red_s[0];
    __syncthreads();
    float av = 0.f;
    if (tid < TT) { av = __expf(e_s[tid] - mx); red_s[tid] = av; }
    __syncthreads();
    for (int off = 64; off > 0; off >>= 1) {
        if (tid < off) red_s[tid] += red_s[tid + off];
        __syncthreads();
    }
    const float inv_sum = __builtin_amdgcn_rcpf(red_s[0]);
    __syncthreads();
    if (tid < TT) e_s[tid] = av * inv_sum;   // alpha
    __syncthreads();

    // --- context: thread tid owns d = 2*tid, 2*tid+1 (one u32 per row) ---
    float acc0 = 0.f, acc1 = 0.f;
    const unsigned* base32 = (const unsigned*)(bH16 + (size_t)b * TT * DD);
    for (int t = 0; t < TT; ++t) {
        const float al = e_s[t];
        const unsigned q = base32[(size_t)t * (DD / 2) + tid];
        acc0 += al * bf16_to_f32(q & 0xFFFFu);
        acc1 += al * bf16_to_f32(q >> 16);
    }
    unsigned short* xr = xcat + (size_t)b * XK;
    ((unsigned*)xr)[tid] = pack_bf16x2(acc0, acc1);

    // --- onehot(text[b,s]) into xcat[b, D : D+C] ---
    if (tid < CC) {
        const int code = text[b * SS + s];
        xr[DD + tid] = (tid == code) ? (unsigned short)0x3F80 : (unsigned short)0;
    }
}

// ---------------------------------------------------------------------------
// LSTM pointwise (gates already include b_ih + b_hh via GEMM bias)
// ---------------------------------------------------------------------------
__global__ __launch_bounds__(256)
void lstm_cell_kernel(const float* __restrict__ gates,   // [B, 4H]
                      float* __restrict__ cstate,        // [B, H]
                      unsigned short* __restrict__ xcat, // [B, XK]
                      unsigned short* __restrict__ hs16, // [B*S, H]
                      int s) {
    const int idx = blockIdx.x * 256 + threadIdx.x;  // B*H threads
    if (idx >= BB * HH) return;
    const int b = idx >> 9;
    const int j = idx & (HH - 1);
    const float* g = gates + (size_t)b * (4 * HH);
    const float ig = fast_sigmoid(g[j]);
    const float fg = fast_sigmoid(g[HH + j]);
    const float gg = fast_tanh(g[2 * HH + j]);
    const float og = fast_sigmoid(g[3 * HH + j]);
    const float cn = fg * cstate[idx] + ig * gg;
    cstate[idx] = cn;
    const float hn = og * fast_tanh(cn);
    const unsigned short hb = f32_to_bf16(hn);
    xcat[(size_t)b * XK + DD + CC + j] = hb;
    hs16[((size_t)b * SS + s) * HH + j] = hb;
}

// ---------------------------------------------------------------------------
// Prologue converters
// ---------------------------------------------------------------------------
__global__ void cvt_f32_bf16_x4_kernel(const float4* __restrict__ src,
                                       uint2* __restrict__ dst, int n4) {
    const int i = blockIdx.x * 256 + threadIdx.x;
    if (i < n4) {
        const float4 v = src[i];
        uint2 o;
        o.x = pack_bf16x2(v.x, v.y);
        o.y = pack_bf16x2(v.z, v.w);
        dst[i] = o;
    }
}

// Wcat[4H, XK] = [ W_ih[4H, D+C] | W_hh[4H, H] ]   (bf16)
__global__ void build_wcat_kernel(const float* __restrict__ W_ih,
                                  const float* __restrict__ W_hh,
                                  unsigned short* __restrict__ Wcat) {
    const int i = blockIdx.x * 256 + threadIdx.x;
    const int total = 4 * HH * XK;
    if (i >= total) return;
    const int r = i / XK, c = i % XK;
    const float v = (c < DD + CC) ? W_ih[(size_t)r * (DD + CC) + c]
                                  : W_hh[(size_t)r * HH + (c - DD - CC)];
    Wcat[i] = f32_to_bf16(v);
}

__global__ void build_bias4h_kernel(const float* __restrict__ b_ih,
                                    const float* __restrict__ b_hh,
                                    float* __restrict__ out) {
    const int i = blockIdx.x * 256 + threadIdx.x;
    if (i < 4 * HH) out[i] = b_ih[i] + b_hh[i];
}

// ---------------------------------------------------------------------------
// Host orchestration (graph-capture safe)
// ---------------------------------------------------------------------------
extern "C" void kernel_launch(void* const* d_in, const int* in_sizes, int n_in,
                              void* d_out, int out_size, void* d_ws, size_t ws_size,
                              hipStream_t stream) {
    (void)in_sizes; (void)n_in; (void)out_size; (void)ws_size;

    const float* batch_H = (const float*)d_in[0];
    const int*   text    = (const int*)d_in[1];
    const float* W_i2h   = (const float*)d_in[2];
    const float* W_h2h   = (const float*)d_in[3];
    const float* b_h2h   = (const float*)d_in[4];
    const float* w_score = (const float*)d_in[5];
    const float* W_ih    = (const float*)d_in[6];
    const float* W_hh    = (const float*)d_in[7];
    const float* b_ih    = (const float*)d_in[8];
    const float* b_hh    = (const float*)d_in[9];
    const float* W_gen   = (const float*)d_in[10];
    const float* b_gen   = (const float*)d_in[11];
    float* probs = (float*)d_out;

    // ---- workspace carve (~161 MB) ----
    char* p = (char*)d_ws;
    auto carve = [&](size_t bytes) -> void* {
        void* r = (void*)p;
        p += (bytes + 255) & ~(size_t)255;
        return r;
    };
    unsigned short* bH16    = (unsigned short*)carve((size_t)BB * TT * DD * 2);
    unsigned short* Hproj16 = (unsigned short*)carve((size_t)BB * TT * HH * 2);
    unsigned short* Wi2h16  = (unsigned short*)carve((size_t)HH * DD * 2);
    unsigned short* Wh2h16  = (unsigned short*)carve((size_t)HH * HH * 2);
    unsigned short* Wcat16  = (unsigned short*)carve((size_t)4 * HH * XK * 2);
    unsigned short* Wgen16  = (unsigned short*)carve((size_t)CC * HH * 2);
    float*          bias4h  = (float*)carve((size_t)4 * HH * 4);
    unsigned short* xcat    = (unsigned short*)carve((size_t)BB * XK * 2);
    float*          hp      = (float*)carve((size_t)BB * HH * 4);
    float*          gates   = (float*)carve((size_t)BB * 4 * HH * 4);
    float*          cstate  = (float*)carve((size_t)BB * HH * 4);
    unsigned short* hs16    = (unsigned short*)carve((size_t)BB * SS * HH * 2);

    hipMemsetAsync(xcat,   0, (size_t)BB * XK * 2, stream);   // h = 0
    hipMemsetAsync(cstate, 0, (size_t)BB * HH * 4, stream);   // c = 0

    // ---- prologue converts (vectorized x4) ----
    auto cvt4 = [&](const float* s_, unsigned short* d_, int n) {
        const int n4 = n / 4;
        cvt_f32_bf16_x4_kernel<<<(n4 + 255) / 256, 256, 0, stream>>>(
            (const float4*)s_, (uint2*)d_, n4);
    };
    cvt4(batch_H, bH16, BB * TT * DD);
    cvt4(W_i2h,   Wi2h16, HH * DD);
    cvt4(W_h2h,   Wh2h16, HH * HH);
    cvt4(W_gen,   Wgen16, CC * HH);
    build_wcat_kernel<<<(4 * HH * XK + 255) / 256, 256, 0, stream>>>(W_ih, W_hh, Wcat16);
    build_bias4h_kernel<<<(4 * HH + 255) / 256, 256, 0, stream>>>(b_ih, b_hh, bias4h);

    // ---- H_proj = batch_H @ W_i2h^T -> bf16 [B*T, H] ----
    {
        dim3 grid(HH / 128, (BB * TT) / 128);
        gemm_bf16_wmma_kernel<<<grid, 256, 0, stream>>>(
            bH16, DD, Wi2h16, DD,
            nullptr, 0, Hproj16, HH, nullptr,
            BB * TT, HH, DD);
    }

    // ---- sequential decode ----
    for (int s = 0; s < SS; ++s) {
        // hp = h @ W_h2h^T + b_h2h   (h lives inside xcat at offset D+C)
        {
            dim3 grid(HH / 128, BB / 128);
            gemm_bf16_wmma_kernel<<<grid, 256, 0, stream>>>(
                xcat + (DD + CC), XK, Wh2h16, HH,
                hp, HH, nullptr, 0, b_h2h,
                BB, HH, HH);
        }
        attn_ctx_kernel<<<BB, 256, 0, stream>>>(Hproj16, bH16, hp, w_score,
                                                text, xcat, s);
        // gates = xcat @ Wcat^T + (b_ih + b_hh)
        {
            dim3 grid((4 * HH) / 128, BB / 128);
            gemm_bf16_wmma_kernel<<<grid, 256, 0, stream>>>(
                xcat, XK, Wcat16, XK,
                gates, 4 * HH, nullptr, 0, bias4h,
                BB, 4 * HH, XK);
        }
        lstm_cell_kernel<<<(BB * HH) / 256, 256, 0, stream>>>(gates, cstate,
                                                              xcat, hs16, s);
    }

    // ---- generator: probs[B*S, C] = hs @ W_gen^T + b_gen ----
    {
        dim3 grid(1 /* ceil(96/128) */, (BB * SS) / 128);
        gemm_bf16_wmma_kernel<<<grid, 256, 0, stream>>>(
            hs16, HH, Wgen16, HH,
            probs, CC, nullptr, 0, b_gen,
            BB * SS, CC, HH);
    }
}
